// HyperbolicMultiHeadAttention_71201967833762
// MI455X (gfx1250) — matrix operand order repaired and checked by reading, not compiled, not run on hardware
//
#include <hip/hip_runtime.h>
#include <cmath>

// ---------- types ----------
typedef __bf16 bhalf;
typedef __attribute__((ext_vector_type(16))) __bf16 v16bf;
typedef __attribute__((ext_vector_type(8)))  __bf16 v8bf;
typedef __attribute__((ext_vector_type(8)))  float  v8f;

#define EPSF 1e-5f

__device__ __forceinline__ bhalf tobf(float f) { return (bhalf)f; }

__device__ __forceinline__ float softplusf(float x) {
    return (x > 20.f) ? x : log1pf(__expf(x));
}

// D = A(16x32) * B(32x16) + C, bf16 inputs, f32 accumulate
__device__ __forceinline__ v8f wmma_bf16(v16bf a, v16bf b, v8f c) {
    return __builtin_amdgcn_wmma_f32_16x16x32_bf16(
        /*neg_a=*/false, a, /*neg_b=*/false, b,
        /*c_mod=*/(short)0, c, /*reuse_a=*/false, /*reuse_b=*/false);
}

// ---- CDNA5 async global->LDS copy (16B per lane), tracked by ASYNCcnt ----
__device__ __forceinline__ void async_ld16(unsigned lds_off, const bhalf* gptr) {
    asm volatile("global_load_async_to_lds_b128 %0, %1, off"
                 :: "v"(lds_off), "v"((unsigned long long)(size_t)gptr)
                 : "memory");
}
__device__ __forceinline__ void wait_async0() {
    asm volatile("s_wait_asynccnt 0x0" ::: "memory");
}

__device__ __forceinline__ v16bf combine16(v8bf lo, v8bf hi) {
    v16bf r;
#pragma unroll
    for (int i = 0; i < 8; ++i) { r[i] = lo[i]; r[i + 8] = hi[i]; }
    return r;
}

// A fragment from global: A is MxK row-major (K contiguous), rows [row0,row0+16), K [k0,k0+32)
__device__ __forceinline__ v16bf load_frag_a(const bhalf* __restrict__ src, int ld,
                                             int row0, int k0, int lane) {
    const int m    = lane & 15;
    const int koff = (lane & 16) ? 8 : 0;
    const bhalf* p = src + (size_t)(row0 + m) * ld + k0 + koff;
    return combine16(*(const v8bf*)p, *(const v8bf*)(p + 16));
}

// B fragment from global: pass Bt (NxK row-major), cols [n0,n0+16), K [k0,k0+32)
__device__ __forceinline__ v16bf load_frag_b(const bhalf* __restrict__ src, int ld,
                                             int n0, int k0, int lane) {
    const int n    = lane & 15;
    const int koff = (lane & 16) ? 8 : 0;
    const bhalf* p = src + (size_t)(n0 + n) * ld + k0 + koff;
    return combine16(*(const v8bf*)p, *(const v8bf*)(p + 16));
}

// ---------- f32 -> bf16 conversion ----------
__global__ void cvt_bf16_kernel(const float* __restrict__ src, bhalf* __restrict__ dst, int n) {
    int i = blockIdx.x * blockDim.x + threadIdx.x;
    int stride = gridDim.x * blockDim.x;
    for (; i < n; i += stride) dst[i] = tobf(src[i]);
}

// ---------- 128x64-tile GEMM with double-buffered async-LDS staging ----------
// OUT = X(2048x1024) @ W^T(1024x1024), one 64-col head block per blockIdx.y.
// grid = (16, 16), block = 256 (8 waves); wave w owns rows [w*16, w*16+16).
// mode 0: exp_map epilogue -> bf16 outb (flat [2048,1024]) + nsq[(b*16+h)*1024+s]
// mode 1: V path -> transposed bf16 outb: vT[((b*16+h)*64+d)*1024 + s]
// mode 2: plain f32 output to outf (flat [2048,1024])
__global__ __launch_bounds__(256) void proj128x64_kernel(
    const bhalf* __restrict__ X, const bhalf* __restrict__ W,
    bhalf* __restrict__ outb, float* __restrict__ outf, float* __restrict__ nsq,
    const float* __restrict__ logc_p, int mode)
{
    __shared__ __align__(16) bhalf lX[2][128 * 32];  // 2 x 8 KB
    __shared__ __align__(16) bhalf lW[2][64 * 32];   // 2 x 4 KB

    const int tid  = threadIdx.x;
    const int wave = tid >> 5;
    const int lane = tid & 31;
    const int rowB = blockIdx.x * 128;
    const int h    = blockIdx.y;
    const int n0   = h * 64;

    // cooperative async stage of one K=32 slab (X: 128x32, W: 64x32)
    auto stage = [&](int buf, int k0) {
        int c = tid;  // X tile: 512 x 16B chunks, 2 per thread
#pragma unroll
        for (int rep = 0; rep < 2; ++rep, c += 256) {
            const int row = c >> 2, kc = c & 3;
            async_ld16((unsigned)(size_t)&lX[buf][c * 8],
                       X + (size_t)(rowB + row) * 1024 + k0 + kc * 8);
        }
        {   // W tile: 256 x 16B chunks, 1 per thread
            const int row = tid >> 2, kc = tid & 3;
            async_ld16((unsigned)(size_t)&lW[buf][tid * 8],
                       W + (size_t)(n0 + row) * 1024 + k0 + kc * 8);
        }
    };

    v8f acc[4];
#pragma unroll
    for (int t = 0; t < 4; ++t)
#pragma unroll
        for (int j = 0; j < 8; ++j) acc[t][j] = 0.f;

    const int m     = lane & 15;
    const int koffA = (lane & 16) ? 8 : 0;
    const int koffB = (lane & 16) ? 8 : 0;

    stage(0, 0);
    for (int it = 0; it < 32; ++it) {
        const int buf = it & 1;
        wait_async0();        // our slab (staged last iteration / prologue) is in LDS
        __syncthreads();      // everyone's slab portion is visible
        if (it + 1 < 32) stage(buf ^ 1, (it + 1) * 32);  // overlap DMA with WMMA

        // hoist ALL fragment loads so ds_loads clause together with one wait
        const bhalf* pa = &lX[buf][(wave * 16 + m) * 32 + koffA];
        v16bf afrag = combine16(*(const v8bf*)pa, *(const v8bf*)(pa + 16));

        v16bf bfrag[4];
#pragma unroll
        for (int t = 0; t < 4; ++t) {
            const bhalf* pb = &lW[buf][(t * 16 + m) * 32 + koffB];
            bfrag[t] = combine16(*(const v8bf*)pb, *(const v8bf*)(pb + 16));
        }
#pragma unroll
        for (int t = 0; t < 4; ++t) acc[t] = wmma_bf16(afrag, bfrag[t], acc[t]);

        __syncthreads();      // done reading buf before it is re-staged
    }

    // ---- epilogues ----
    const int row0 = rowB + wave * 16;
    const int half = (lane >> 4) & 1;
    const int col  = lane & 15;
    const int bidx = row0 >> 10;   // batch index; 128-row blocks never straddle S=1024
    const int sbas = row0 & 1023;

    if (mode == 2) {
#pragma unroll
        for (int t = 0; t < 4; ++t)
#pragma unroll
            for (int j = 0; j < 8; ++j)
                outf[(size_t)(row0 + j + 8 * half) * 1024 + n0 + t * 16 + col] = acc[t][j];
        return;
    }
    if (mode == 1) {
        const size_t base = (size_t)(bidx * 16 + h) * 64 * 1024;
#pragma unroll
        for (int t = 0; t < 4; ++t) {
            const int d = t * 16 + col;
#pragma unroll
            for (int j = 0; j < 8; ++j)
                outb[base + (size_t)d * 1024 + (sbas + j + 8 * half)] = tobf(acc[t][j]);
        }
        return;
    }

    // mode 0: exp_map_at_origin rescale + squared-norm output
    const float cc = softplusf(*logc_p);
    const float sc = sqrtf(cc);

    float p2[8];
#pragma unroll
    for (int j = 0; j < 8; ++j) {
        float s = 0.f;
#pragma unroll
        for (int t = 0; t < 4; ++t) s += acc[t][j] * acc[t][j];
#pragma unroll
        for (int mask = 1; mask < 16; mask <<= 1) s += __shfl_xor(s, mask, 32);
        p2[j] = s;  // row (j + 8*half) squared norm, replicated across the 16-lane half
    }
    float alpha[8];
#pragma unroll
    for (int j = 0; j < 8; ++j) {
        const float vn = fmaxf(sqrtf(p2[j]), EPSF);
        alpha[j] = tanhf(sc * vn) / (sc * vn);
    }
#pragma unroll
    for (int t = 0; t < 4; ++t)
#pragma unroll
        for (int j = 0; j < 8; ++j)
            outb[(size_t)(row0 + j + 8 * half) * 1024 + n0 + t * 16 + col] =
                tobf(alpha[j] * acc[t][j]);
#pragma unroll
    for (int j = 0; j < 8; ++j)
        if (col == j)
            nsq[(size_t)(bidx * 16 + h) * 1024 + (sbas + j + 8 * half)] =
                alpha[j] * alpha[j] * p2[j];
}

// ---------- scores + softmax -> bf16 attention weights ----------
// grid = (S/16, B*H), block = 256 (8 waves). Each wave: 16 rows x 128 cols (8 WMMA tiles).
__global__ __launch_bounds__(256) void attn_scores_kernel(
    const bhalf* __restrict__ qh, const bhalf* __restrict__ kh,
    const float* __restrict__ xns, const float* __restrict__ yns,
    bhalf* __restrict__ wts,
    const float* __restrict__ logc_p, const float* __restrict__ beta_p,
    const float* __restrict__ bias_p)
{
    const int tid  = threadIdx.x;
    const int wave = tid >> 5;
    const int lane = tid & 31;
    const int r0   = blockIdx.x * 16;
    const int bh   = blockIdx.y;
    const int b    = bh >> 4;
    const int h    = bh & 15;

    const bhalf* qbase = qh + (size_t)(b * 1024 + r0) * 1024 + h * 64;
    const bhalf* kbase = kh + (size_t)(b * 1024) * 1024 + h * 64;
    const float* xr    = xns + (size_t)bh * 1024;
    const float* yr    = yns + (size_t)bh * 1024;

    v8f acc[8];
#pragma unroll
    for (int t = 0; t < 8; ++t)
#pragma unroll
        for (int j = 0; j < 8; ++j) acc[t][j] = 0.f;

#pragma unroll
    for (int k0 = 0; k0 < 64; k0 += 32) {
        v16bf a = load_frag_a(qbase, 1024, 0, k0, lane);
#pragma unroll
        for (int t = 0; t < 8; ++t) {
            v16bf bfrag = load_frag_b(kbase, 1024, wave * 128 + t * 16, k0, lane);
            acc[t] = wmma_bf16(a, bfrag, acc[t]);
        }
    }

    const float cc   = softplusf(*logc_p);
    const float sc   = sqrtf(cc);
    const float rsc  = 1.f / sc;
    const float beta = softplusf(*beta_p);
    const float bias = *bias_p;

    const int half = (lane >> 4) & 1;
    const int col  = lane & 15;

    float xq[8];
#pragma unroll
    for (int j = 0; j < 8; ++j) xq[j] = xr[r0 + j + 8 * half];

#pragma unroll
    for (int t = 0; t < 8; ++t) {
        const float yv  = yr[wave * 128 + t * 16 + col];
        const float oyv = 1.f - cc * yv;
#pragma unroll
        for (int j = 0; j < 8; ++j) {
            const float xy    = acc[t][j];
            const float dsq   = xq[j] - 2.f * xy + yv;
            const float denom = fmaxf((1.f - cc * xq[j]) * oyv, EPSF);
            const float arg   = fmaxf(1.f + 2.f * cc * dsq / denom, 1.f + EPSF);
            const float dist  = rsc * logf(arg + sqrtf(arg * arg - 1.f));  // arccosh
            acc[t][j] = -beta * dist - bias;
        }
    }

    // ---- row softmax over 1024 cols (cross-wave via LDS) ----
    __shared__ float red[8][16];
    __shared__ float gred[16];

#pragma unroll
    for (int j = 0; j < 8; ++j) {
        float mx = -3.4e38f;
#pragma unroll
        for (int t = 0; t < 8; ++t) mx = fmaxf(mx, acc[t][j]);
#pragma unroll
        for (int mask = 1; mask < 16; mask <<= 1) mx = fmaxf(mx, __shfl_xor(mx, mask, 32));
        if (col == j) red[wave][j + 8 * half] = mx;
    }
    __syncthreads();
    if (tid < 16) {
        float mx = -3.4e38f;
#pragma unroll
        for (int w = 0; w < 8; ++w) mx = fmaxf(mx, red[w][tid]);
        gred[tid] = mx;
    }
    __syncthreads();
    float gm[8];
#pragma unroll
    for (int j = 0; j < 8; ++j) gm[j] = gred[j + 8 * half];
    __syncthreads();   // gred consumed; safe to reuse red/gred below

    // sum phase: predicated LDS store only, single barrier after the loop
#pragma unroll
    for (int j = 0; j < 8; ++j) {
        float s = 0.f;
#pragma unroll
        for (int t = 0; t < 8; ++t) {
            const float e = __expf(acc[t][j] - gm[j]);
            acc[t][j] = e;
            s += e;
        }
#pragma unroll
        for (int mask = 1; mask < 16; mask <<= 1) s += __shfl_xor(s, mask, 32);
        if (col == j) red[wave][j + 8 * half] = s;
    }
    __syncthreads();
    if (tid < 16) {
        float s = 0.f;
#pragma unroll
        for (int w = 0; w < 8; ++w) s += red[w][tid];
        gred[tid] = s;
    }
    __syncthreads();

    const size_t wbase = ((size_t)bh * 1024 + r0) * 1024;
#pragma unroll
    for (int j = 0; j < 8; ++j) {
        const float inv = 1.f / gred[j + 8 * half];
#pragma unroll
        for (int t = 0; t < 8; ++t)
            wts[wbase + (size_t)(j + 8 * half) * 1024 + wave * 128 + t * 16 + col] =
                tobf(acc[t][j] * inv);
    }
}

// ---------- out_tan = weights @ v_tan (per head) -> concat bf16 [B,S,D] ----------
// grid = (S/16, B*H), block = 32.
__global__ __launch_bounds__(32) void attn_av_kernel(
    const bhalf* __restrict__ wts, const bhalf* __restrict__ vT,
    bhalf* __restrict__ concat)
{
    const int lane = threadIdx.x & 31;
    const int r0   = blockIdx.x * 16;
    const int bh   = blockIdx.y;
    const int b    = bh >> 4;
    const int h    = bh & 15;

    const bhalf* A  = wts + ((size_t)bh * 1024 + r0) * 1024;  // 16 x 1024, K contiguous
    const bhalf* Bv = vT + (size_t)bh * 64 * 1024;            // 64 rows (d) x 1024 (t)

    v8f acc[4];
#pragma unroll
    for (int t = 0; t < 4; ++t)
#pragma unroll
        for (int j = 0; j < 8; ++j) acc[t][j] = 0.f;

    for (int k0 = 0; k0 < 1024; k0 += 32) {
        v16bf a = load_frag_a(A, 1024, 0, k0, lane);
        v16bf bfrag[4];
#pragma unroll
        for (int t = 0; t < 4; ++t) bfrag[t] = load_frag_b(Bv, 1024, t * 16, k0, lane);
#pragma unroll
        for (int t = 0; t < 4; ++t) acc[t] = wmma_bf16(a, bfrag[t], acc[t]);
    }

    const int half = (lane >> 4) & 1;
    const int col  = lane & 15;
#pragma unroll
    for (int t = 0; t < 4; ++t)
#pragma unroll
        for (int j = 0; j < 8; ++j)
            concat[(size_t)(b * 1024 + r0 + j + 8 * half) * 1024 + h * 64 + t * 16 + col] =
                tobf(acc[t][j]);
}

// ---------- host-side orchestration ----------
extern "C" void kernel_launch(void* const* d_in, const int* in_sizes, int n_in,
                              void* d_out, int out_size, void* d_ws, size_t ws_size,
                              hipStream_t stream) {
    const float* x    = (const float*)d_in[0];  // [2,1024,1024]
    const float* W_q  = (const float*)d_in[1];
    const float* W_k  = (const float*)d_in[2];
    const float* W_v  = (const float*)d_in[3];
    const float* W_o  = (const float*)d_in[4];
    const float* logc = (const float*)d_in[5];
    const float* beta = (const float*)d_in[6];
    const float* bias = (const float*)d_in[7];
    float* out = (float*)d_out;

    const int NX = 2 * 1024 * 1024;
    const int NW = 1024 * 1024;

    char* p = (char*)d_ws;
    size_t off = 0;
    auto carve = [&](size_t bytes) {
        void* r = p + off;
        off += (bytes + 255) & ~(size_t)255;
        return r;
    };
    bhalf* xb     = (bhalf*)carve((size_t)NX * 2);
    bhalf* Wqb    = (bhalf*)carve((size_t)NW * 2);
    bhalf* Wkb    = (bhalf*)carve((size_t)NW * 2);
    bhalf* Wvb    = (bhalf*)carve((size_t)NW * 2);
    bhalf* Wob    = (bhalf*)carve((size_t)NW * 2);
    bhalf* qh     = (bhalf*)carve((size_t)NX * 2);
    bhalf* kh     = (bhalf*)carve((size_t)NX * 2);
    bhalf* vT     = (bhalf*)carve((size_t)NX * 2);
    bhalf* concat = (bhalf*)carve((size_t)NX * 2);
    float* xns    = (float*)carve((size_t)32 * 1024 * 4);
    float* yns    = (float*)carve((size_t)32 * 1024 * 4);
    bhalf* wts    = (bhalf*)carve((size_t)2 * 16 * 1024 * 1024 * 2);  // 64 MB
    (void)ws_size; (void)in_sizes; (void)n_in; (void)out_size;

    // 1) f32 -> bf16
    cvt_bf16_kernel<<<2048, 256, 0, stream>>>(x,   xb,  NX);
    cvt_bf16_kernel<<<1024, 256, 0, stream>>>(W_q, Wqb, NW);
    cvt_bf16_kernel<<<1024, 256, 0, stream>>>(W_k, Wkb, NW);
    cvt_bf16_kernel<<<1024, 256, 0, stream>>>(W_v, Wvb, NW);
    cvt_bf16_kernel<<<1024, 256, 0, stream>>>(W_o, Wob, NW);

    // 2) projections (+ exp_map epilogue for Q/K, transposed store for V)
    dim3 gproj(16, 16);
    proj128x64_kernel<<<gproj, 256, 0, stream>>>(xb, Wqb, qh, nullptr, xns, logc, 0);
    proj128x64_kernel<<<gproj, 256, 0, stream>>>(xb, Wkb, kh, nullptr, yns, logc, 0);
    proj128x64_kernel<<<gproj, 256, 0, stream>>>(xb, Wvb, vT, nullptr, nullptr, logc, 1);

    // 3) Poincare distance + softmax -> bf16 weights
    attn_scores_kernel<<<dim3(64, 32), 256, 0, stream>>>(qh, kh, xns, yns, wts,
                                                         logc, beta, bias);

    // 4) weights @ v_tan -> concat
    attn_av_kernel<<<dim3(64, 32), 32, 0, stream>>>(wts, vT, concat);

    // 5) concat @ W_o^T -> f32 output
    proj128x64_kernel<<<gproj, 256, 0, stream>>>(concat, Wob, nullptr, out, nullptr, logc, 2);
}
